// ScaledDotProductAttention_51238959841839
// MI455X (gfx1250) — matrix operand order repaired and checked by reading, not compiled, not run on hardware
//
#include <hip/hip_runtime.h>

// CDNA5 / gfx1250 causal attention (with the reference's 1e-9 mask quirk),
// bf16 WMMA compute, fp32 accumulate, writes both out and attention_weights.
//
// v3: skip WMMA/K-loads for fully-masked key tiles (constant-fill 1e-9 region),
//     vectorize softmax sweeps (b64 LDS, b128 nontemporal weights stores),
//     vector LDS reads for GEMM2 P fragments.

typedef __attribute__((ext_vector_type(16))) __bf16 v16bf;
typedef __attribute__((ext_vector_type(8)))  __bf16 v8bf;
typedef __attribute__((ext_vector_type(4)))  __bf16 v4bf;
typedef __attribute__((ext_vector_type(8)))  float  v8f;
typedef __attribute__((ext_vector_type(4)))  float  v4f;

#define S_LEN 2048
#define D_LEN 64

__device__ __forceinline__ __bf16 f2bf(float f) { return (__bf16)f; }

__global__ __launch_bounds__(32)
void attn_wmma_bf16_kernel(const float* __restrict__ Q,
                           const float* __restrict__ K,
                           const float* __restrict__ V,
                           float* __restrict__ Out,
                           float* __restrict__ W)
{
    __shared__ __bf16 slab[16 * S_LEN];   // 64 KB: 16 q-rows x 2048 scores/probs (bf16)

    const int lane = threadIdx.x;         // wave32
    const int n    = lane & 15;
    const int half = lane >> 4;

    const int bh = blockIdx.x >> 7;       // 24 (b*h)
    const int qt = blockIdx.x & 127;      // 128 q-tiles of 16 rows
    const int q0 = qt * 16;

    const float* __restrict__ Qb = Q + (size_t)bh * S_LEN * D_LEN;
    const float* __restrict__ Kb = K + (size_t)bh * S_LEN * D_LEN;
    const float* __restrict__ Vb = V + (size_t)bh * S_LEN * D_LEN;
    float* __restrict__ Wb  = W   + (size_t)bh * S_LEN * S_LEN;
    float* __restrict__ Ob  = Out + (size_t)bh * S_LEN * D_LEN;

    // ---- Load Q A-fragments (16x32 bf16 each, k-chunks d[0:32) and d[32:64)) ----
    // A layout (16-bit, 16x32): lane row = lane&15; element e -> k = (e<8?e:e+8) + 8*half
    v16bf a0, a1;
    {
        const float* qp = Qb + (size_t)(q0 + n) * D_LEN;
        #pragma unroll
        for (int e = 0; e < 16; ++e) {
            int d0 = ((e < 8) ? e : e + 8) + 8 * half;
            a0[e] = f2bf(qp[d0]);
            a1[e] = f2bf(qp[d0 + 32]);
        }
    }

    // Every row of this q-tile has masked columns iff qt < 127 (plus the diagonal
    // tile's own partial masking, handled by the select below). Fold the 1e-9
    // mask constant into the running max analytically.
    float rmax[8];
    #pragma unroll
    for (int v = 0; v < 8; ++v) rmax[v] = (qt < 127) ? 1e-9f : -3.0e38f;

    // ---- GEMM1 (unmasked tiles j < qt): scores = (Q K^T)/8, spill bf16 ----
    for (int j = 0; j < qt; ++j) {
        // B layout (16-bit, 32x16): lane column = key; element e -> k(d) = (e<8?e:e+8) + 8*half
        const float* kp = Kb + (size_t)(j * 16 + n) * D_LEN;
        __builtin_prefetch(kp + 16 * D_LEN, 0, 0);        // next K tile (stream)
        v16bf b0, b1;
        #pragma unroll
        for (int e = 0; e < 16; ++e) {
            int dd = ((e < 8) ? e : e + 8) + 8 * half;
            b0[e] = f2bf(kp[dd]);
            b1[e] = f2bf(kp[dd + 32]);
        }
        v8f c = {};
        c = __builtin_amdgcn_wmma_f32_16x16x32_bf16(false, a0, false, b0, (short)0, c, false, false);
        c = __builtin_amdgcn_wmma_f32_16x16x32_bf16(false, a1, false, b1, (short)0, c, false, false);

        // C/D layout: lane n holds column (key) n; VGPR v holds row m = v + 8*half
        #pragma unroll
        for (int v = 0; v < 8; ++v) {
            const int row = v + 8 * half;
            float s = c[v] * 0.125f;                      // 1/sqrt(64)
            rmax[v] = fmaxf(rmax[v], s);
            slab[row * S_LEN + j * 16 + n] = f2bf(s);
        }
    }

    // ---- Diagonal tile (j == qt): apply the 1e-9 mask quirk per element ----
    {
        const int key = q0 + n;
        const float* kp = Kb + (size_t)key * D_LEN;
        v16bf b0, b1;
        #pragma unroll
        for (int e = 0; e < 16; ++e) {
            int dd = ((e < 8) ? e : e + 8) + 8 * half;
            b0[e] = f2bf(kp[dd]);
            b1[e] = f2bf(kp[dd + 32]);
        }
        v8f c = {};
        c = __builtin_amdgcn_wmma_f32_16x16x32_bf16(false, a0, false, b0, (short)0, c, false, false);
        c = __builtin_amdgcn_wmma_f32_16x16x32_bf16(false, a1, false, b1, (short)0, c, false, false);
        #pragma unroll
        for (int v = 0; v < 8; ++v) {
            const int row = v + 8 * half;
            float s = c[v] * 0.125f;
            s = (key > q0 + row) ? 1e-9f : s;
            rmax[v] = fmaxf(rmax[v], s);
            slab[row * S_LEN + qt * 16 + n] = f2bf(s);
        }
    }

    // ---- Fully-masked region (cols >= (qt+1)*16): constant 1e-9, no WMMA ----
    {
        const int fillStart = (qt + 1) * 16;
        const int nChunks   = (S_LEN - fillStart) >> 3;   // 8 bf16 = 16B per chunk
        v8bf mv;
        #pragma unroll
        for (int e = 0; e < 8; ++e) mv[e] = f2bf(1e-9f);
        for (int r = 0; r < 16; ++r) {
            v8bf* dst = (v8bf*)&slab[r * S_LEN + fillStart];
            for (int idx = lane; idx < nChunks; idx += 32) dst[idx] = mv;
        }
    }

    // One cross-lane reduce per row group (columns covered by the 16-lane union).
    #pragma unroll
    for (int v = 0; v < 8; ++v) {
        float t = rmax[v];
        t = fmaxf(t, __shfl_xor(t, 1, 32));
        t = fmaxf(t, __shfl_xor(t, 2, 32));
        t = fmaxf(t, __shfl_xor(t, 4, 32));
        t = fmaxf(t, __shfl_xor(t, 8, 32));
        rmax[v] = t;                                      // replicated within each half
    }

    // ---- Softmax (quirk semantics): p = exp(s - max); Z = sum p; w = p/Z ----
    float invZ[16];
    #pragma unroll
    for (int r = 0; r < 16; ++r) {
        const float mr = __shfl(rmax[r & 7], (r >> 3) * 16, 32);
        float ps = 0.0f;
        for (int it = 0; it < S_LEN / 128; ++it) {        // 4 cols per lane per step
            const int c0 = (it * 32 + lane) * 4;
            v4bf sv = *(const v4bf*)&slab[r * S_LEN + c0];
            float p0 = __expf((float)sv[0] - mr);
            float p1 = __expf((float)sv[1] - mr);
            float p2 = __expf((float)sv[2] - mr);
            float p3 = __expf((float)sv[3] - mr);
            ps += (p0 + p1) + (p2 + p3);
            v4bf pv;
            pv[0] = f2bf(p0); pv[1] = f2bf(p1); pv[2] = f2bf(p2); pv[3] = f2bf(p3);
            *(v4bf*)&slab[r * S_LEN + c0] = pv;           // keep unnormalized p for GEMM2
        }
        ps += __shfl_xor(ps, 1, 32);
        ps += __shfl_xor(ps, 2, 32);
        ps += __shfl_xor(ps, 4, 32);
        ps += __shfl_xor(ps, 8, 32);
        ps += __shfl_xor(ps, 16, 32);
        const float iz = 1.0f / ps;
        invZ[r] = iz;
        // dominant 402 MB stream: b128 nontemporal stores (don't thrash L2)
        float* wrow = Wb + (size_t)(q0 + r) * S_LEN;
        for (int it = 0; it < S_LEN / 128; ++it) {
            const int c0 = (it * 32 + lane) * 4;
            v4bf pv = *(const v4bf*)&slab[r * S_LEN + c0];
            v4f w4;
            w4.x = (float)pv[0] * iz;
            w4.y = (float)pv[1] * iz;
            w4.z = (float)pv[2] * iz;
            w4.w = (float)pv[3] * iz;
            __builtin_nontemporal_store(w4, (v4f*)&wrow[c0]);
        }
    }

    float myinv[8];
    #pragma unroll
    for (int v = 0; v < 8; ++v) myinv[v] = half ? invZ[v + 8] : invZ[v];

    // ---- GEMM2: out = (P V) * invZ, P read as bf16 A-fragments from LDS ----
    v8f acc[4] = {v8f{}, v8f{}, v8f{}, v8f{}};
    for (int j = 0; j < S_LEN / 32; ++j) {
        const __bf16* prow = &slab[n * S_LEN + j * 32 + 8 * half];
        v8bf lo = *(const v8bf*)prow;                     // k run [.., ..+8)
        v8bf hi = *(const v8bf*)(prow + 16);              // k run [..+16, ..+24)
        v16bf pa;
        #pragma unroll
        for (int e = 0; e < 8; ++e) { pa[e] = lo[e]; pa[e + 8] = hi[e]; }
        __builtin_prefetch(Vb + (size_t)(j * 32 + 32 + 16 * half) * D_LEN + n, 0, 0); // next V tile
        #pragma unroll
        for (int t = 0; t < 4; ++t) {
            v16bf vb;                                     // B: lane column = d index
            const int d = t * 16 + n;
            #pragma unroll
            for (int e = 0; e < 16; ++e) {
                int key = j * 32 + ((e < 8) ? e : e + 8) + 8 * half;
                vb[e] = f2bf(Vb[(size_t)key * D_LEN + d]);
            }
            acc[t] = __builtin_amdgcn_wmma_f32_16x16x32_bf16(false, pa, false, vb,
                                                             (short)0, acc[t], false, false);
        }
    }

    #pragma unroll
    for (int t = 0; t < 4; ++t) {
        #pragma unroll
        for (int v = 0; v < 8; ++v) {
            const int row = v + 8 * half;
            Ob[(size_t)(q0 + row) * D_LEN + t * 16 + n] = acc[t][v] * myinv[v];
        }
    }
}

extern "C" void kernel_launch(void* const* d_in, const int* in_sizes, int n_in,
                              void* d_out, int out_size, void* d_ws, size_t ws_size,
                              hipStream_t stream) {
    (void)in_sizes; (void)n_in; (void)d_ws; (void)ws_size; (void)out_size;
    const float* Q = (const float*)d_in[0];
    const float* K = (const float*)d_in[1];
    const float* V = (const float*)d_in[2];
    // d_in[3] is the tril mask; causality is applied analytically in-kernel.

    const int B = 2, H = 12;
    float* out = (float*)d_out;                           // [B,H,S,D]
    float* wts = out + (size_t)B * H * S_LEN * D_LEN;     // [B,H,S,S]

    dim3 grid(B * H * (S_LEN / 16));                      // 3072 workgroups
    dim3 block(32);                                       // one wave32 each
    attn_wmma_bf16_kernel<<<grid, block, 0, stream>>>(Q, K, V, out, wts);
}
